// TransformerBlock_63591285784764
// MI455X (gfx1250) — matrix operand order repaired and checked by reading, not compiled, not run on hardware
//
#include <hip/hip_runtime.h>
#include <hip/hip_bf16.h>
#include <cstdint>

// ---------------------------------------------------------------------------
// MI455X (gfx1250) transformer block: bf16 WMMA GEMMs + flash attention.
// wave32, v_wmma_f32_16x16x32_bf16, f32 accumulate.
// Tile staging via Tensor Data Mover (tensor_load_to_lds) with descriptor
// padding producing a bank-conflict-free 72-element LDS pitch; double-buffered
// so the TDM overlaps the WMMA stream. s_wait_tensorcnt gates consumption.
// ---------------------------------------------------------------------------

#define B_  4
#define S_  2048
#define D_  512
#define H_  8
#define F_  2048
#define HD_ 64
#define M_  (B_ * S_)     // 8192 rows

#define AP 72             // LDS pitch (elements) for 64-wide K tiles

typedef __attribute__((ext_vector_type(16))) __bf16 v16bf;
typedef __attribute__((ext_vector_type(8)))  float  v8f;
typedef __attribute__((ext_vector_type(4)))  unsigned int u32x4;
typedef __attribute__((ext_vector_type(8)))  int    i32x8;
typedef __attribute__((ext_vector_type(4)))  int    i32x4;

union FragB {
    uint4 u[2];
    v16bf v;
};

__device__ __forceinline__ uint16_t f2bf(float f) {
    uint32_t u = __float_as_uint(f);
    uint32_t r = u + 0x7FFFu + ((u >> 16) & 1u);   // round-to-nearest-even
    return (uint16_t)(r >> 16);
}

// ---------------------------------------------------------------------------
// TDM: async 2D tile load (64 bf16 wide, tile_h rows) global -> LDS.
// LDS padding: after every 32 DWORDs (one 128B row) insert 4 DWORDs (16B),
// i.e. hardware-applied 72-element pitch. Tracked by TENSORcnt.
// ---------------------------------------------------------------------------
__device__ __forceinline__ void tdm_load_tile64(uint32_t lds_addr,
                                                const void* gptr,
                                                uint32_t tile_h,
                                                uint32_t stride_elems) {
    uint64_t ga = (uint64_t)(uintptr_t)gptr;
    u32x4 g0;
    g0[0] = 1u;                                  // count=1, user descriptor
    g0[1] = lds_addr;                            // LDS byte address
    g0[2] = (uint32_t)ga;                        // global_addr[31:0]
    g0[3] = (uint32_t)(ga >> 32) | (2u << 30);   // global_addr[56:32] | type=2
    i32x8 g1;
    g1[0] = (int)((1u << 16)                     // data_size = 2 bytes
                | (1u << 20)                     // pad_enable
                | (4u << 22)                     // pad_interval: 32 DWORDs
                | (3u << 25));                   // pad_amount: 4 DWORDs
    g1[1] = (int)(64u << 16);                    // tensor_dim0 = 64
    g1[2] = (int)(tile_h << 16);                 // tensor_dim1 = tile_h
    g1[3] = (int)(64u << 16);                    // tile_dim0 = 64
    g1[4] = (int)tile_h;                         // tile_dim1 = tile_h, tile_dim2=0
    g1[5] = (int)stride_elems;                   // tensor_dim0_stride[31:0]
    g1[6] = 0;
    g1[7] = 0;
    i32x4 z4 = {0, 0, 0, 0};
#if __has_include(<hip/amd_detail/amd_gfx1250_TDM.h>)
    i32x8 z8 = {0, 0, 0, 0, 0, 0, 0, 0};
    __builtin_amdgcn_tensor_load_to_lds(g0, g1, z4, z4, z8, 0);
#elif __has_builtin(__builtin_amdgcn_tensor_load_to_lds)
    __builtin_amdgcn_tensor_load_to_lds(g0, g1, z4, z4, 0);
#else
    asm volatile("tensor_load_to_lds %0, %1" :: "s"(g0), "s"(g1) : "memory");
#endif
}

__device__ __forceinline__ uint32_t lds_off(const void* p) {
    return (uint32_t)(uintptr_t)p;               // low 32 bits = LDS offset
}

// ---------------------------------------------------------------------------
// Prep kernels
// ---------------------------------------------------------------------------
__global__ void cvt_bf16_kernel(const float* __restrict__ in,
                                uint16_t* __restrict__ out, int n) {
    int i = blockIdx.x * 256 + threadIdx.x;
    if (i < n) out[i] = f2bf(in[i]);
}

// W[K][N] fp32 (row-major) -> Wt[N][K] bf16
__global__ void transpose_cvt_kernel(const float* __restrict__ W,
                                     uint16_t* __restrict__ Wt,
                                     int K, int N) {
    int idx = blockIdx.x * 256 + threadIdx.x;
    if (idx < K * N) {
        int n = idx / K;
        int k = idx - n * K;
        Wt[idx] = f2bf(W[(size_t)k * N + n]);
    }
}

// ---------------------------------------------------------------------------
// Generic bf16 GEMM:  C[M_,N] = A[M_,K] * Bt[N,K]^T  + epilogue
// 128x128 tile per 256-thread workgroup (8 waves, each 32x64).
// Double-buffered TDM pipeline: each wave DMAs its own 16-row slices.
// EPI: 0 = QKV split (q scaled, v transposed), 1 = o-proj + resid (fp32 out),
//      2 = GELU -> bf16, 3 = ffn2 + resid (fp32 out)
// ---------------------------------------------------------------------------
template <int EPI>
__global__ __launch_bounds__(256)
void gemm_bf16_kernel(const uint16_t* __restrict__ A,
                      const uint16_t* __restrict__ Bt,
                      const float* __restrict__ bias,
                      const float* __restrict__ resid,
                      float* __restrict__ outf,
                      uint16_t* __restrict__ outb,
                      uint16_t* __restrict__ qout,
                      uint16_t* __restrict__ kout,
                      uint16_t* __restrict__ vout,
                      int K, int N) {
    __shared__ __align__(16) uint16_t As[2][128 * AP];
    __shared__ __align__(16) uint16_t Bs[2][128 * AP];

    const int tid  = threadIdx.x;
    const int lane = tid & 31;
    const int w    = tid >> 5;
    const int half = lane >> 4;
    const int ln   = lane & 15;
    const int wr   = w & 3;        // row block of 32
    const int wc   = w >> 2;       // col block of 64
    const int mbase = blockIdx.x * 128;
    const int nbase = blockIdx.y * 128;

    const uint32_t asb = lds_off(&As[0][0]) + (uint32_t)(w * 16 * AP * 2);
    const uint32_t bsb = lds_off(&Bs[0][0]) + (uint32_t)(w * 16 * AP * 2);
    const uint32_t BUFB = 128 * AP * 2;    // bytes per buffer

    const uint16_t* Ag  = A  + (size_t)(mbase + w * 16) * K;
    const uint16_t* Btg = Bt + (size_t)(nbase + w * 16) * K;

    v8f acc[2][4] = {};

    // prologue: DMA first tiles
    tdm_load_tile64(asb, Ag, 16, (uint32_t)K);
    tdm_load_tile64(bsb, Btg, 16, (uint32_t)K);
    __builtin_amdgcn_s_wait_tensorcnt(0);
    __syncthreads();

    int buf = 0;
    for (int kb = 0; kb < K; kb += 64) {
        // DMA next tiles into the other buffer while we compute
        if (kb + 64 < K) {
            tdm_load_tile64(asb + (buf ^ 1) * BUFB, Ag + kb + 64, 16, (uint32_t)K);
            tdm_load_tile64(bsb + (buf ^ 1) * BUFB, Btg + kb + 64, 16, (uint32_t)K);
        }
        const uint16_t* Asl = As[buf];
        const uint16_t* Bsl = Bs[buf];

#pragma unroll
        for (int kk = 0; kk < 64; kk += 32) {
            FragB a[2];
#pragma unroll
            for (int i = 0; i < 2; i++) {
                int r = wr * 32 + i * 16 + ln;
                a[i].u[0] = *(const uint4*)&Asl[r * AP + kk + 8 * half];
                a[i].u[1] = *(const uint4*)&Asl[r * AP + kk + 16 + 8 * half];
            }
#pragma unroll
            for (int j = 0; j < 4; j++) {
                FragB b;
                int n = wc * 64 + j * 16 + ln;
                b.u[0] = *(const uint4*)&Bsl[n * AP + kk + 8 * half];
                b.u[1] = *(const uint4*)&Bsl[n * AP + kk + 16 + 8 * half];
#pragma unroll
                for (int i = 0; i < 2; i++)
                    acc[i][j] = __builtin_amdgcn_wmma_f32_16x16x32_bf16(
                        false, a[i].v, false, b.v, (short)0, acc[i][j], false, false);
            }
        }
        __builtin_amdgcn_s_wait_tensorcnt(0);   // my next-tile DMAs done
        __syncthreads();                         // everyone done + all DMAs done
        buf ^= 1;
    }

    // ---- epilogue ----
#pragma unroll
    for (int i = 0; i < 2; i++) {
#pragma unroll
        for (int j = 0; j < 4; j++) {
#pragma unroll
            for (int r = 0; r < 8; r++) {
                int m = mbase + wr * 32 + i * 16 + r + 8 * half;
                int n = nbase + wc * 64 + j * 16 + ln;
                float val = acc[i][j][r] + bias[n];
                if (EPI == 0) {
                    int b = m >> 11, s = m & (S_ - 1);
                    if (n < D_) {
                        int h = n >> 6, hd = n & 63;
                        qout[(((size_t)(b * H_ + h)) * S_ + s) * HD_ + hd] =
                            f2bf(val * 0.125f);                         // *SCALE
                    } else if (n < 2 * D_) {
                        int d = n - D_;
                        int h = d >> 6, hd = d & 63;
                        kout[(((size_t)(b * H_ + h)) * S_ + s) * HD_ + hd] = f2bf(val);
                    } else {
                        int d = n - 2 * D_;
                        int h = d >> 6, hd = d & 63;
                        vout[(((size_t)(b * H_ + h)) * HD_ + hd) * S_ + s] = f2bf(val);
                    }
                } else if (EPI == 1) {
                    outf[(size_t)m * N + n] = val + resid[(size_t)m * N + n];
                } else if (EPI == 2) {
                    float g = 0.5f * val * (1.0f + erff(val * 0.70710678118654752f));
                    outb[(size_t)m * N + n] = f2bf(g);
                } else {
                    outf[(size_t)m * N + n] = val + resid[(size_t)m * N + n];
                }
            }
        }
    }
}

// ---------------------------------------------------------------------------
// Flash attention: one workgroup = 128 query rows of one (b,h).
// Each of 8 waves owns 16 query rows. 64-key chunks, online softmax.
// K/V^T chunks staged by TDM, double-buffered (8-row slice per wave).
// q,k: [B,H,S,HD] bf16 (q pre-scaled). vt: [B,H,HD,S] bf16.
// o: [B,S,D] bf16 (heads merged).
// ---------------------------------------------------------------------------
__global__ __launch_bounds__(256)
void attn_kernel(const uint16_t* __restrict__ q,
                 const uint16_t* __restrict__ k,
                 const uint16_t* __restrict__ vt,
                 uint16_t* __restrict__ o) {
    __shared__ __align__(16) uint16_t Ks[2][64 * AP];     // [key][hd]
    __shared__ __align__(16) uint16_t Vts[2][64 * AP];    // [hd][key]
    __shared__ __align__(16) uint16_t Ps[8][16 * AP];     // per-wave P tile

    const int bh   = blockIdx.y;
    const int qb   = blockIdx.x * 128;
    const int tid  = threadIdx.x;
    const int lane = tid & 31;
    const int w    = tid >> 5;
    const int half = lane >> 4;
    const int ln   = lane & 15;

    const uint32_t ksb  = lds_off(&Ks[0][0])  + (uint32_t)(w * 8 * AP * 2);
    const uint32_t vtsb = lds_off(&Vts[0][0]) + (uint32_t)(w * 8 * AP * 2);
    const uint32_t BUFB = 64 * AP * 2;

    const uint16_t* kg  = k  + ((size_t)bh * S_ + w * 8) * HD_;   // rows of K
    const uint16_t* vtg = vt + ((size_t)bh * HD_ + w * 8) * S_;   // rows of V^T

    // Q fragments (A layout), loaded once directly from global
    FragB aq[2];
    const uint16_t* qrow = q + ((size_t)bh * S_ + qb + w * 16 + ln) * HD_;
#pragma unroll
    for (int f = 0; f < 2; f++) {
        aq[f].u[0] = *(const uint4*)(qrow + f * 32 + 8 * half);
        aq[f].u[1] = *(const uint4*)(qrow + f * 32 + 16 + 8 * half);
    }

    float m_i[8], l_i[8];
#pragma unroll
    for (int r = 0; r < 8; r++) { m_i[r] = -1e30f; l_i[r] = 0.0f; }
    v8f oacc[4] = {};

    // prologue: DMA first K/V^T chunk
    tdm_load_tile64(ksb, kg, 8, HD_);
    tdm_load_tile64(vtsb, vtg, 8, S_);
    __builtin_amdgcn_s_wait_tensorcnt(0);
    __syncthreads();

    int buf = 0;
    for (int kc = 0; kc < S_; kc += 64) {
        if (kc + 64 < S_) {
            tdm_load_tile64(ksb + (buf ^ 1) * BUFB, kg + (size_t)(kc + 64) * HD_, 8, HD_);
            tdm_load_tile64(vtsb + (buf ^ 1) * BUFB, vtg + kc + 64, 8, S_);
        }
        const uint16_t* Ksl  = Ks[buf];
        const uint16_t* Vtsl = Vts[buf];

        // ---- S = Q @ K^T for this wave's 16 rows x 64 keys ----
        v8f sacc[4] = {};
#pragma unroll
        for (int j = 0; j < 4; j++) {
#pragma unroll
            for (int f = 0; f < 2; f++) {
                FragB bk;
                int rr = j * 16 + ln;
                bk.u[0] = *(const uint4*)&Ksl[rr * AP + f * 32 + 8 * half];
                bk.u[1] = *(const uint4*)&Ksl[rr * AP + f * 32 + 16 + 8 * half];
                sacc[j] = __builtin_amdgcn_wmma_f32_16x16x32_bf16(
                    false, aq[f].v, false, bk.v, (short)0, sacc[j], false, false);
            }
        }

        // ---- online softmax update (row = r + 8*half) ----
#pragma unroll
        for (int r = 0; r < 8; r++) {
            float mx = sacc[0][r];
#pragma unroll
            for (int j = 1; j < 4; j++) mx = fmaxf(mx, sacc[j][r]);
#pragma unroll
            for (int off = 1; off < 16; off <<= 1)
                mx = fmaxf(mx, __shfl_xor(mx, off, 32));
            float newm = fmaxf(m_i[r], mx);
            float corr = __expf(m_i[r] - newm);
            float rs = 0.0f;
#pragma unroll
            for (int j = 0; j < 4; j++) {
                float pj = __expf(sacc[j][r] - newm);
                sacc[j][r] = pj;
                rs += pj;
            }
#pragma unroll
            for (int off = 1; off < 16; off <<= 1)
                rs += __shfl_xor(rs, off, 32);
            l_i[r] = l_i[r] * corr + rs;
            m_i[r] = newm;
#pragma unroll
            for (int j = 0; j < 4; j++) oacc[j][r] *= corr;
        }

        // ---- re-layout P (C layout -> A layout) via per-wave LDS tile ----
#pragma unroll
        for (int j = 0; j < 4; j++)
#pragma unroll
            for (int r = 0; r < 8; r++)
                Ps[w][(r + 8 * half) * AP + j * 16 + ln] = f2bf(sacc[j][r]);
        asm volatile("s_wait_dscnt 0" ::: "memory");   // wave-local LDS RAW

        // ---- O += P @ V ----
#pragma unroll
        for (int kk = 0; kk < 2; kk++) {
            FragB ap;
            ap.u[0] = *(const uint4*)&Ps[w][ln * AP + kk * 32 + 8 * half];
            ap.u[1] = *(const uint4*)&Ps[w][ln * AP + kk * 32 + 16 + 8 * half];
#pragma unroll
            for (int j = 0; j < 4; j++) {
                FragB bv;
                int vr = j * 16 + ln;
                bv.u[0] = *(const uint4*)&Vtsl[vr * AP + kk * 32 + 8 * half];
                bv.u[1] = *(const uint4*)&Vtsl[vr * AP + kk * 32 + 16 + 8 * half];
                oacc[j] = __builtin_amdgcn_wmma_f32_16x16x32_bf16(
                    false, ap.v, false, bv.v, (short)0, oacc[j], false, false);
            }
        }
        __builtin_amdgcn_s_wait_tensorcnt(0);
        __syncthreads();
        buf ^= 1;
    }

    // ---- normalize + store merged-head output ----
    const int b = bh >> 3, h = bh & 7;
#pragma unroll
    for (int j = 0; j < 4; j++) {
#pragma unroll
        for (int r = 0; r < 8; r++) {
            float val = oacc[j][r] / l_i[r];
            int row = qb + w * 16 + r + 8 * half;
            o[((size_t)(b * S_ + row)) * D_ + h * HD_ + j * 16 + ln] = f2bf(val);
        }
    }
}

// ---------------------------------------------------------------------------
// LayerNorm over D=512: one block (128 threads) per row.
// Writes fp32 (residual / output) and optionally bf16 (next GEMM operand).
// ---------------------------------------------------------------------------
__global__ __launch_bounds__(128)
void ln_kernel(const float* __restrict__ in,
               const float* __restrict__ g,
               const float* __restrict__ be,
               float* __restrict__ outf,
               uint16_t* __restrict__ outb) {
    const int row = blockIdx.x;
    const int tid = threadIdx.x;
    const float* p = in + (size_t)row * D_;

    float v[4], s = 0.0f, s2 = 0.0f;
#pragma unroll
    for (int i = 0; i < 4; i++) {
        v[i] = p[tid + i * 128];
        s += v[i];
        s2 += v[i] * v[i];
    }
#pragma unroll
    for (int off = 16; off; off >>= 1) {
        s  += __shfl_xor(s, off, 32);
        s2 += __shfl_xor(s2, off, 32);
    }
    __shared__ float rs[4], rq[4];
    if ((tid & 31) == 0) { rs[tid >> 5] = s; rq[tid >> 5] = s2; }
    __syncthreads();
    s  = rs[0] + rs[1] + rs[2] + rs[3];
    s2 = rq[0] + rq[1] + rq[2] + rq[3];

    const float mu  = s * (1.0f / D_);
    const float var = s2 * (1.0f / D_) - mu * mu;
    const float inv = rsqrtf(var + 1e-5f);
#pragma unroll
    for (int i = 0; i < 4; i++) {
        int c = tid + i * 128;
        float y = (v[i] - mu) * inv * g[c] + be[c];
        outf[(size_t)row * D_ + c] = y;
        if (outb) outb[(size_t)row * D_ + c] = f2bf(y);
    }
}

// ---------------------------------------------------------------------------
// Launcher
// ---------------------------------------------------------------------------
extern "C" void kernel_launch(void* const* d_in, const int* in_sizes, int n_in,
                              void* d_out, int out_size, void* d_ws, size_t ws_size,
                              hipStream_t stream) {
    const float* x     = (const float*)d_in[0];
    const float* w_qkv = (const float*)d_in[1];
    const float* b_qkv = (const float*)d_in[2];
    const float* w_o   = (const float*)d_in[3];
    const float* b_o   = (const float*)d_in[4];
    const float* w1    = (const float*)d_in[5];
    const float* b1    = (const float*)d_in[6];
    const float* w2    = (const float*)d_in[7];
    const float* b2    = (const float*)d_in[8];
    const float* g1    = (const float*)d_in[9];
    const float* be1   = (const float*)d_in[10];
    const float* g2    = (const float*)d_in[11];
    const float* be2   = (const float*)d_in[12];

    uint8_t* p = (uint8_t*)d_ws;
    auto carve = [&](size_t bytes) -> void* {
        void* r = (void*)p;
        p += (bytes + 255) & ~(size_t)255;
        return r;
    };
    uint16_t* xb    = (uint16_t*)carve((size_t)M_ * D_ * 2);
    uint16_t* wqkvT = (uint16_t*)carve((size_t)3 * D_ * D_ * 2);
    uint16_t* woT   = (uint16_t*)carve((size_t)D_ * D_ * 2);
    uint16_t* w1T   = (uint16_t*)carve((size_t)D_ * F_ * 2);
    uint16_t* w2T   = (uint16_t*)carve((size_t)F_ * D_ * 2);
    uint16_t* qb    = (uint16_t*)carve((size_t)M_ * D_ * 2);
    uint16_t* kb    = (uint16_t*)carve((size_t)M_ * D_ * 2);
    uint16_t* vtb   = (uint16_t*)carve((size_t)M_ * D_ * 2);
    uint16_t* attb  = (uint16_t*)carve((size_t)M_ * D_ * 2);
    float*    pre1  = (float*)   carve((size_t)M_ * D_ * 4);
    float*    h1f   = (float*)   carve((size_t)M_ * D_ * 4);
    uint16_t* h1b   = (uint16_t*)carve((size_t)M_ * D_ * 2);
    uint16_t* gelb  = (uint16_t*)carve((size_t)M_ * F_ * 2);
    float*    pre2  = (float*)   carve((size_t)M_ * D_ * 4);

    // prep: convert / transpose
    cvt_bf16_kernel<<<(M_ * D_ + 255) / 256, 256, 0, stream>>>(x, xb, M_ * D_);
    transpose_cvt_kernel<<<(D_ * 3 * D_ + 255) / 256, 256, 0, stream>>>(w_qkv, wqkvT, D_, 3 * D_);
    transpose_cvt_kernel<<<(D_ * D_ + 255) / 256, 256, 0, stream>>>(w_o, woT, D_, D_);
    transpose_cvt_kernel<<<(D_ * F_ + 255) / 256, 256, 0, stream>>>(w1, w1T, D_, F_);
    transpose_cvt_kernel<<<(F_ * D_ + 255) / 256, 256, 0, stream>>>(w2, w2T, F_, D_);

    // qkv = x @ w_qkv + b_qkv  (split into q/k/v^T, q scaled)
    gemm_bf16_kernel<0><<<dim3(M_ / 128, (3 * D_) / 128), 256, 0, stream>>>(
        xb, wqkvT, b_qkv, nullptr, nullptr, nullptr, qb, kb, vtb, D_, 3 * D_);

    // attention
    attn_kernel<<<dim3(S_ / 128, B_ * H_), 256, 0, stream>>>(qb, kb, vtb, attb);

    // o-proj + residual
    gemm_bf16_kernel<1><<<dim3(M_ / 128, D_ / 128), 256, 0, stream>>>(
        attb, woT, b_o, x, pre1, nullptr, nullptr, nullptr, nullptr, D_, D_);

    // LN1
    ln_kernel<<<M_, 128, 0, stream>>>(pre1, g1, be1, h1f, h1b);

    // ffn1 + gelu
    gemm_bf16_kernel<2><<<dim3(M_ / 128, F_ / 128), 256, 0, stream>>>(
        h1b, w1T, b1, nullptr, nullptr, gelb, nullptr, nullptr, nullptr, D_, F_);

    // ffn2 + residual
    gemm_bf16_kernel<3><<<dim3(M_ / 128, D_ / 128), 256, 0, stream>>>(
        gelb, w2T, b2, h1f, pre2, nullptr, nullptr, nullptr, nullptr, F_, D_);

    // LN2 -> output
    ln_kernel<<<M_, 128, 0, stream>>>(pre2, g2, be2, (float*)d_out, nullptr);
}
